// DeformableConv2d_33767032881836
// MI455X (gfx1250) — compile-verified, matrix operand order checked by reading, not run on hardware
//
#include <hip/hip_runtime.h>
#include <hip/hip_bf16.h>
#include <math.h>

typedef __attribute__((ext_vector_type(16))) _Float16 v16h;
typedef __attribute__((ext_vector_type(8)))  float    v8f;

#define Bsz 4
#define C_IN 128
#define O_OUT 128
#define HW 64
#define K2 9
#define KTOT 1152                 // C_IN * K2
#define NKC 36                    // KTOT / 32
#define WPA_HALVES (8 * NKC * 32 * 16)   // org_w A-fragments: 147456 f16
#define WPB_HALVES (2 * NKC * 32 * 16)   // offset/mask A-fragments: 36864 f16

// kmap (ISA 7.12.2, 16-bit A/B 16x32 fragment): lane<16 -> K {0..7,16..23},
// lane>=16 -> K {8..15,24..31}; per-lane 16 halves contiguous (32 B).
__device__ __forceinline__ int kmap_fwd(int l, int e) {
  return (l < 16) ? ((e < 8) ? e : e + 8) : ((e < 8) ? e + 8 : e + 16);
}
__device__ __forceinline__ void kmap_inv(int koff, int n, int& l, int& e) {
  if      (koff < 8)  { l = n;      e = koff;      }
  else if (koff < 16) { l = n + 16; e = koff - 8;  }
  else if (koff < 24) { l = n;      e = koff - 8;  }
  else                { l = n + 16; e = koff - 16; }
}

// ---------------------------------------------------------------------------
// Kernel 1: pack BOTH weight matrices (f32) into f16 WMMA A-fragments.
//   wpackA: org_w [128][128][3][3] -> 8 M-tiles x 36 K-chunks
//   wpackB: [offset_w;mask_w] as 32x1152 (rows 27..31 zero) -> 2 M-tiles
// ---------------------------------------------------------------------------
__global__ __launch_bounds__(256) void dcn_pack(const float* __restrict__ orgw,
                                                const float* __restrict__ offw,
                                                const float* __restrict__ maskw,
                                                _Float16* __restrict__ wpackA,
                                                _Float16* __restrict__ wpackB) {
  int i = blockIdx.x * 256 + threadIdx.x;
  if (i >= WPA_HALVES + WPB_HALVES) return;
  const bool isB = (i >= WPA_HALVES);
  const int j = isB ? (i - WPA_HALVES) : i;
  const int e  = j & 15;
  const int l  = (j >> 4) & 31;
  const int kc = (j >> 9) % NKC;
  const int mt = (j >> 9) / NKC;
  const int K  = (kc << 5) + kmap_fwd(l, e);   // flat contraction = c*9 + k
  const int M  = (mt << 4) + (l & 15);
  const int c  = K / 9;
  const int k  = K - 9 * c;
  float w;
  if (!isB) {
    w = orgw[((M << 7) + c) * 9 + k];
    wpackA[j] = (_Float16)w;
  } else {
    if      (M < 18) w = offw [((M << 7) + c) * 9 + k];
    else if (M < 27) w = maskw[(((M - 18) << 7) + c) * 9 + k];
    else             w = 0.f;
    wpackB[j] = (_Float16)w;
  }
}

// ---------------------------------------------------------------------------
// Kernel 2: offset/mask convs as WMMA GEMM (M=32[27], K=1152, N=16/WG).
// All 256 threads build the im2col B tile (fragment layout) in LDS, then
// waves 0..1 run 36 chained v_wmma each; bias + sigmoid applied at store.
// om layout: [B][27][4096] f32; ch<18 offset, ch>=18 sigmoid(mask).
// ---------------------------------------------------------------------------
__global__ __launch_bounds__(256) void dcn_offmask_wmma(const float* __restrict__ x,
                                                        const _Float16* __restrict__ wpackB,
                                                        const float* __restrict__ offb,
                                                        const float* __restrict__ maskb,
                                                        float* __restrict__ om) {
  __shared__ alignas(32) _Float16 tile[KTOT * 16];  // 36 KB B-fragments
  const int tid = threadIdx.x;
  const int blk = blockIdx.x;
  const int b   = blk >> 8;
  const int posbase = (blk & 255) << 4;

  // ---- im2col build: element (K=c*9+k, n) = x[b,c,ho-1+ki,wo-1+kj] --------
  const float* xb = x + ((size_t)b << 19);
  const int n0 = tid & 15;
  for (int it = tid; it < KTOT * 16; it += 256) {
    const int K  = it >> 4;
    const int c  = K / 9;
    const int k  = K - 9 * c;
    const int ki = k / 3, kj = k % 3;
    const int p  = posbase + n0;
    const int ho = p >> 6, wo = p & 63;
    const int row = ho - 1 + ki;
    const int col = wo - 1 + kj;
    float v = 0.f;
    if ((unsigned)row < 64u && (unsigned)col < 64u)
      v = xb[((size_t)c << 12) + (row << 6) + col];
    int l, e;
    kmap_inv(K & 31, n0, l, e);
    tile[((((K >> 5) << 5) + l) << 4) + e] = (_Float16)v;
  }
  __syncthreads();

  // ---- 2-wave WMMA GEMM (uniform branch; EXEC all ones inside) ------------
  const int wv = tid >> 5;
  if (wv < 2) {
    const int lane = tid & 31;
    const _Float16* ap = wpackB + ((((size_t)wv * NKC << 5) + lane) << 4);
    const _Float16* bp = &tile[lane << 4];
    v8f acc = {};
    for (int kc = 0; kc < NKC; ++kc) {
      if (kc + 1 < NKC) __builtin_prefetch(ap + ((kc + 1) << 9), 0, 1);
      v16h A  = *(const v16h*)(ap + (kc << 9));
      v16h Bf = *(const v16h*)(bp + (kc << 9));
      acc = __builtin_amdgcn_wmma_f32_16x16x32_f16(false, A, false, Bf,
                                                   (short)0, acc, false, false);
    }
    const int n = lane & 15;
    const int mbase = (wv << 4) + ((lane >> 4) << 3);
#pragma unroll
    for (int r = 0; r < 8; ++r) {
      const int ch = mbase + r;
      if (ch < 27) {
        float val = acc[r] + ((ch < 18) ? offb[ch] : maskb[ch - 18]);
        if (ch >= 18) val = 1.f / (1.f + __expf(-val));   // sigmoid
        om[(((size_t)(b * 27 + ch)) << 12) + posbase + n] = val;
      }
    }
  }
}

// ---------------------------------------------------------------------------
// Kernel 3: fused bilinear sampling + GEMM via v_wmma_f32_16x16x32_f16.
// Block = 256 thr (8 waves); owns batch b and 16 spatial positions.
// Wave w computes output channels [16w,16w+16): D = sum_36 A(16x32)B(32x16).
// ---------------------------------------------------------------------------
__global__ __launch_bounds__(256) void dcn_main(const float* __restrict__ x,
                                                const _Float16* __restrict__ wpackA,
                                                const float* __restrict__ om,
                                                float* __restrict__ out) {
  __shared__ alignas(32) _Float16 valLDS[KTOT * 16]; // 36 KB B-fragments
  __shared__ int4   idx4[144];
  __shared__ float4 wgt4[144];

  const int tid = threadIdx.x;
  const int blk = blockIdx.x;
  const int b   = blk >> 8;
  const int posbase = (blk & 255) << 4;

  // ---- Phase A: bilinear params for 16 positions x 9 taps -----------------
  if (tid < 144) {
    const int n = tid & 15, k = tid >> 4;
    const int p  = posbase + n;
    const int ho = p >> 6, wo = p & 63;
    const float* omb = om + (((size_t)b * 27) << 12);
    const float dy = omb[((size_t)(2 * k)     << 12) + p];
    const float dx = omb[((size_t)(2 * k + 1) << 12) + p];
    const float m  = omb[((size_t)(18 + k)    << 12) + p];
    const float py = (float)(ho - 1 + k / 3) + dy;
    const float px = (float)(wo - 1 + k % 3) + dx;
    const float y0f = floorf(py), x0f = floorf(px);
    const float wy = py - y0f, wx = px - x0f;
    const int y0 = (int)y0f, x0 = (int)x0f, y1 = y0 + 1, x1 = x0 + 1;
    const float vy0 = ((unsigned)y0 < 64u) ? 1.f : 0.f;
    const float vy1 = ((unsigned)y1 < 64u) ? 1.f : 0.f;
    const float vx0 = ((unsigned)x0 < 64u) ? 1.f : 0.f;
    const float vx1 = ((unsigned)x1 < 64u) ? 1.f : 0.f;
    const int y0c = min(max(y0, 0), 63), y1c = min(max(y1, 0), 63);
    const int x0c = min(max(x0, 0), 63), x1c = min(max(x1, 0), 63);
    idx4[tid] = make_int4(y0c * 64 + x0c, y0c * 64 + x1c,
                          y1c * 64 + x0c, y1c * 64 + x1c);
    wgt4[tid] = make_float4((1.f - wy) * (1.f - wx) * m * vy0 * vx0,
                            (1.f - wy) * wx         * m * vy0 * vx1,
                            wy * (1.f - wx)         * m * vy1 * vx0,
                            wy * wx                 * m * vy1 * vx1);
  }
  __syncthreads();

  // ---- Phase B: gather/blend -> f16 B-fragments in LDS --------------------
  const float* xb = x + ((size_t)b << 19);
  const int n0 = tid & 15;
  for (int it = tid; it < KTOT * 16; it += 256) {
    const int K = it >> 4;
    const int c = K / 9;
    const int k = K - 9 * c;
    const int t = (k << 4) | n0;
    const int4   id = idx4[t];
    const float4 w  = wgt4[t];
    const float* xp = xb + ((size_t)c << 12);
    const float v = w.x * xp[id.x] + w.y * xp[id.y] +
                    w.z * xp[id.z] + w.w * xp[id.w];
    int l, e;
    kmap_inv(K & 31, n0, l, e);
    valLDS[((((K >> 5) << 5) + l) << 4) + e] = (_Float16)v;
  }
  __syncthreads();

  // ---- Phase C: per-wave WMMA over 36 K-chunks (EXEC all ones) ------------
  const int lane = tid & 31;
  const int wv   = tid >> 5;
  const _Float16* ap = wpackA + ((((size_t)wv * NKC << 5) + lane) << 4);
  const _Float16* bp = &valLDS[lane << 4];
  v8f acc = {};
  for (int kc = 0; kc < NKC; ++kc) {
    if (kc + 1 < NKC) __builtin_prefetch(ap + ((kc + 1) << 9), 0, 1);
    v16h A  = *(const v16h*)(ap + (kc << 9));   // 2x global_load_b128
    v16h Bf = *(const v16h*)(bp + (kc << 9));   // 2x ds_load_b128
    acc = __builtin_amdgcn_wmma_f32_16x16x32_f16(false, A, false, Bf,
                                                 (short)0, acc, false, false);
  }

  // ---- store D (lane<16: M=r, lane>=16: M=r+8), coalesced over n ----------
  const int n = lane & 15;
  const int mbase = (wv << 4) + ((lane >> 4) << 3);
  float* op = out + ((((size_t)(b << 7) + mbase)) << 12) + posbase + n;
#pragma unroll
  for (int r = 0; r < 8; ++r) op[(size_t)r << 12] = acc[r];
}

// ---------------------------------------------------------------------------
extern "C" void kernel_launch(void* const* d_in, const int* in_sizes, int n_in,
                              void* d_out, int out_size, void* d_ws, size_t ws_size,
                              hipStream_t stream) {
  const float* x        = (const float*)d_in[0];
  const float* org_w    = (const float*)d_in[1];
  const float* offset_w = (const float*)d_in[2];
  const float* offset_b = (const float*)d_in[3];
  const float* mask_w   = (const float*)d_in[4];
  const float* mask_b   = (const float*)d_in[5];
  float* out = (float*)d_out;

  _Float16* wpackA = (_Float16*)d_ws;                                   // 294912 B
  _Float16* wpackB = (_Float16*)((char*)d_ws + WPA_HALVES * 2);         //  73728 B
  float*    om     = (float*)((char*)d_ws + (WPA_HALVES + WPB_HALVES) * 2); // 1769472 B

  const int packN = WPA_HALVES + WPB_HALVES;
  dcn_pack        <<<(packN + 255) / 256, 256, 0, stream>>>(org_w, offset_w, mask_w,
                                                            wpackA, wpackB);
  dcn_offmask_wmma<<<Bsz * (HW * HW / 16), 256, 0, stream>>>(x, wpackB, offset_b,
                                                             mask_b, om);
  dcn_main        <<<Bsz * (HW * HW / 16), 256, 0, stream>>>(x, wpackA, om, out);
}
